// StandardRoIHeadWarper_60541859004651
// MI455X (gfx1250) — matrix operand order repaired and checked by reading, not compiled, not run on hardware
//
#include <hip/hip_runtime.h>

typedef __bf16 bf16;
typedef __attribute__((ext_vector_type(16))) __bf16 v16bf;
typedef __attribute__((ext_vector_type(8)))  __bf16 bf16x8;
typedef __attribute__((ext_vector_type(8)))  float   v8f;
typedef __attribute__((ext_vector_type(4)))  int     v4i;

#define RROIS   1024      // B*N
#define NB      2
#define CCH     256
#define DIN     12544     // 256*49
#define FHID    1024
#define NCLS    80
#define NCLSP   128       // padded 81 -> 128
#define NREG    320
#define NREGP   384       // padded 320 -> 384 (N tiles of 128)
#define MCAND   40960     // 512*80
#define MAXDET  100

// padded LDS tile strides (in bf16 elements) for bank-conflict avoidance
#define ASTRIDE 40        // 64x32 A tile rows, 80B = 20 dwords -> conflict-free
#define BSTRIDE 136       // 32x128 B tile rows, 272B = 68 dwords -> 2-way max

// ---------------------------------------------------------------------------
// CDNA5 async global -> LDS copy (ASYNCcnt-tracked).
// Builtin signature (from hipcc diagnostic): (v4i AS1*, v4i AS3*, imm, imm).
// ---------------------------------------------------------------------------
__device__ __forceinline__ void async_ld_b128(const bf16* g, bf16* l) {
#if __has_builtin(__builtin_amdgcn_global_load_async_to_lds_b128)
    __builtin_amdgcn_global_load_async_to_lds_b128(
        (__attribute__((address_space(1))) v4i*)g,
        (__attribute__((address_space(3))) v4i*)l, 0, 0);
#else
    unsigned lo = (unsigned)(size_t)l;
    asm volatile("global_load_async_to_lds_b128 %0, %1, off"
                 :: "v"(lo), "v"(g) : "memory");
#endif
}

__device__ __forceinline__ void wait_async0() {
#if __has_builtin(__builtin_amdgcn_s_wait_asynccnt)
    __builtin_amdgcn_s_wait_asynccnt(0);
#else
    asm volatile("s_wait_asynccnt 0x0" ::: "memory");
#endif
}

// ---------------------------------------------------------------------------
// fp32 -> bf16 convert with optional column padding (zeros)
// ---------------------------------------------------------------------------
__global__ void cvt_pad_bf16_kernel(const float* __restrict__ src, bf16* __restrict__ dst,
                                    int K, int Nsrc, int Ndst) {
    size_t i = (size_t)blockIdx.x * blockDim.x + threadIdx.x;
    size_t total = (size_t)K * Ndst;
    if (i >= total) return;
    int n = (int)(i % Ndst);
    size_t k = i / Ndst;
    dst[i] = (n < Nsrc) ? (bf16)src[k * (size_t)Nsrc + n] : (bf16)0.0f;
}

// ---------------------------------------------------------------------------
// RoI-Align: one block per RoI, one thread per channel; writes bf16 activation
// matrix x[r][c*49+p] (the reference's transpose(0,2,1).reshape).
// ---------------------------------------------------------------------------
__global__ __launch_bounds__(256) void roi_align_kernel(
    const float* __restrict__ f0, const float* __restrict__ f1,
    const float* __restrict__ f2, const float* __restrict__ f3,
    const float* __restrict__ proposals, bf16* __restrict__ xb)
{
    const int r = blockIdx.x;
    const int c = threadIdx.x;
    const int b = r >> 9;
    const float* box = proposals + (size_t)r * 4;
    float x1 = box[0], y1 = box[1], x2 = box[2], y2 = box[3];
    float scale = sqrtf(fmaxf(x2 - x1, 0.f) * fmaxf(y2 - y1, 0.f));
    int lvl = (int)floorf(log2f(scale * (1.0f / 56.0f) + 1e-6f));
    lvl = lvl < 0 ? 0 : (lvl > 3 ? 3 : lvl);
    const float* feat = (lvl == 0) ? f0 : (lvl == 1) ? f1 : (lvl == 2) ? f2 : f3;
    const int H = 192 >> lvl;
    const float sc = 1.0f / (float)(4 << lvl);
    const float* fc = feat + ((size_t)b * CCH + c) * (size_t)H * H;

    float bx1 = x1 * sc - 0.5f, by1 = y1 * sc - 0.5f;
    float bx2 = x2 * sc - 0.5f, by2 = y2 * sc - 0.5f;
    float bw = (bx2 - bx1) * (1.0f / 7.0f), bh = (by2 - by1) * (1.0f / 7.0f);

    bf16* xo = xb + (size_t)r * DIN + (size_t)c * 49;
    for (int p = 0; p < 49; ++p) {
        int iy = p / 7, ix = p % 7;
        float yg = by1 + bh * ((float)iy + 0.5f);
        float xg = bx1 + bw * ((float)ix + 0.5f);
        bool valid = (yg > -1.0f) && (yg < (float)H) && (xg > -1.0f) && (xg < (float)H);
        float yc = fminf(fmaxf(yg, 0.f), (float)(H - 1));
        float xc = fminf(fmaxf(xg, 0.f), (float)(H - 1));
        float y0f = floorf(yc), x0f = floorf(xc);
        int y0i = (int)y0f, x0i = (int)x0f;
        int y1i = (y0i + 1 < H - 1) ? y0i + 1 : H - 1;
        int x1i = (x0i + 1 < H - 1) ? x0i + 1 : H - 1;
        float ly = yc - y0f, lx = xc - x0f;
        float v00 = fc[(size_t)y0i * H + x0i];
        float v01 = fc[(size_t)y0i * H + x1i];
        float v10 = fc[(size_t)y1i * H + x0i];
        float v11 = fc[(size_t)y1i * H + x1i];
        float v = (1.f - ly) * (1.f - lx) * v00 + (1.f - ly) * lx * v01 +
                  ly * (1.f - lx) * v10 + ly * lx * v11;
        xo[p] = (bf16)(valid ? v : 0.0f);
    }
}

// ---------------------------------------------------------------------------
// WMMA GEMM with async-LDS double buffering.
//   Block (256 thr = 8 waves) computes a 64x128 tile of C = act(A@W + bias).
//   Per 32-K step: stage A(64x32) + B(32x128) bf16 tiles to LDS with
//   GLOBAL_LOAD_ASYNC_TO_LDS_B128 (ASYNCcnt), compute from the other buffer.
//   Waves arranged 4x2; each wave owns a 16x64 strip (A frag reused 4x).
// Fragment layouts per CDNA5 ISA 7.12.2 (wave32):
//   A 16x32 bf16 : lane&15 = M row, lane>>4 = K half; elems 0..7 = K(h*8..),
//                  elems 8..15 = K(16+h*8..)
//   B 32x16 bf16 : lane&15 = N col, lane>>4 = K half; same K order as A
//   D 16x16 f32  : lane&15 = N col, vgpr g -> M row g + 8*(lane>>4)
// ---------------------------------------------------------------------------
__global__ __launch_bounds__(256) void wmma_gemm_bf16_kernel(
    const bf16* __restrict__ A, int lda,
    const bf16* __restrict__ W, int ldw,
    const float* __restrict__ bias, int biasN,
    float* __restrict__ outF, bf16* __restrict__ outB,
    int M, int N, int K, int relu)
{
    __shared__ __align__(16) bf16 Atile[2][64 * ASTRIDE];
    __shared__ __align__(16) bf16 Btile[2][32 * BSTRIDE];

    const int tid  = threadIdx.x;
    const int lane = tid & 31;
    const int wv   = tid >> 5;             // 0..7
    const int wm4  = wv >> 1;              // 0..3 : 16-row group
    const int wn2  = wv & 1;               // 0..1 : 64-col group

    const int nblk = N >> 7;               // 128-wide N tiles
    const int m0b  = (blockIdx.x / nblk) << 6;
    const int n0b  = (blockIdx.x % nblk) << 7;

    const int half = lane >> 4;
    const int mrow = lane & 15;

    // staging assignments (per thread): one A chunk + two B chunks of 16B
    const int arow = tid >> 2, aseg = tid & 3;

    auto stage = [&](int buf, int kb) {
        const bf16* ga = A + (size_t)(m0b + arow) * lda + kb + aseg * 8;
        async_ld_b128(ga, &Atile[buf][arow * ASTRIDE + aseg * 8]);
#pragma unroll
        for (int j = 0; j < 2; ++j) {
            int chunk = tid + 256 * j;
            int brow = chunk >> 4, bseg = chunk & 15;
            const bf16* gb = W + (size_t)(kb + brow) * ldw + n0b + bseg * 8;
            async_ld_b128(gb, &Btile[buf][brow * BSTRIDE + bseg * 8]);
        }
    };

    v8f acc[4] = {v8f{}, v8f{}, v8f{}, v8f{}};

    stage(0, 0);
    wait_async0();
    __syncthreads();

    int buf = 0;
    for (int kb = 0; kb < K; kb += 32) {
        if (kb + 32 < K) stage(buf ^ 1, kb + 32);

        const bf16* ap = &Atile[buf][(wm4 * 16 + mrow) * ASTRIDE + half * 8];
        bf16x8 lo = *(const bf16x8*)(ap);
        bf16x8 hi = *(const bf16x8*)(ap + 16);
        v16bf a;
#pragma unroll
        for (int i = 0; i < 8; ++i) { a[i] = lo[i]; a[8 + i] = hi[i]; }

        const bf16* bp = &Btile[buf][wn2 * 64 + mrow];
        v16bf b0, b1, b2, b3;
#pragma unroll
        for (int i = 0; i < 8; ++i) {
            const bf16* pl = bp + (half * 8 + i) * BSTRIDE;
            const bf16* ph = bp + (16 + half * 8 + i) * BSTRIDE;
            b0[i] = pl[0];  b0[8 + i] = ph[0];
            b1[i] = pl[16]; b1[8 + i] = ph[16];
            b2[i] = pl[32]; b2[8 + i] = ph[32];
            b3[i] = pl[48]; b3[8 + i] = ph[48];
        }

        acc[0] = __builtin_amdgcn_wmma_f32_16x16x32_bf16(false, a, false, b0, (short)0, acc[0], false, false);
        acc[1] = __builtin_amdgcn_wmma_f32_16x16x32_bf16(false, a, false, b1, (short)0, acc[1], false, false);
        acc[2] = __builtin_amdgcn_wmma_f32_16x16x32_bf16(false, a, false, b2, (short)0, acc[2], false, false);
        acc[3] = __builtin_amdgcn_wmma_f32_16x16x32_bf16(false, a, false, b3, (short)0, acc[3], false, false);

        wait_async0();
        __syncthreads();
        buf ^= 1;
    }

#pragma unroll
    for (int t = 0; t < 4; ++t) {
        int ncol = n0b + wn2 * 64 + t * 16 + (lane & 15);
        float bv = (ncol < biasN) ? bias[ncol] : 0.0f;
#pragma unroll
        for (int g = 0; g < 8; ++g) {
            float v = acc[t][g] + bv;
            if (relu) v = fmaxf(v, 0.0f);
            int mr = m0b + wm4 * 16 + g + (half << 3);
            if (outB) outB[(size_t)mr * N + ncol] = (bf16)v;
            else      outF[(size_t)mr * N + ncol] = v;
        }
    }
}

// ---------------------------------------------------------------------------
// Softmax over 81 classes + delta2bbox decode + score threshold per RoI.
// ---------------------------------------------------------------------------
__global__ __launch_bounds__(128) void smax_decode_kernel(
    const float* __restrict__ clsf,      // R x 128 (cols 0..80 valid)
    const float* __restrict__ regf,      // R x 384 (cols 0..319 valid)
    const float* __restrict__ proposals, // R x 4
    float* __restrict__ cand_boxes, float* __restrict__ boxes_off,
    float* __restrict__ s0)
{
    __shared__ float sm[128];
    const int r = blockIdx.x, tid = threadIdx.x;
    float lv = (tid < 81) ? clsf[(size_t)r * NCLSP + tid] : -3.0e38f;
    sm[tid] = lv; __syncthreads();
    for (int s = 64; s > 0; s >>= 1) { if (tid < s) sm[tid] = fmaxf(sm[tid], sm[tid + s]); __syncthreads(); }
    float mx = sm[0]; __syncthreads();
    float e = (tid < 81) ? expf(lv - mx) : 0.0f;
    sm[tid] = e; __syncthreads();
    for (int s = 64; s > 0; s >>= 1) { if (tid < s) sm[tid] += sm[tid + s]; __syncthreads(); }
    float sum = sm[0];

    if (tid < NCLS) {
        float score = e / sum;
        float s0v = (score > 0.05f) ? score : -1.0f;
        const float* box = proposals + (size_t)r * 4;
        float px = (box[0] + box[2]) * 0.5f, py = (box[1] + box[3]) * 0.5f;
        float pw = box[2] - box[0], ph = box[3] - box[1];
        const float* d = regf + (size_t)r * NREGP + tid * 4;
        float dx = d[0] * 0.1f, dy = d[1] * 0.1f;
        float dw = d[2] * 0.2f, dh = d[3] * 0.2f;
        const float MR = 4.135166556742356f;   // log(1000/16)
        dw = fminf(fmaxf(dw, -MR), MR);
        dh = fminf(fmaxf(dh, -MR), MR);
        float gx = px + pw * dx, gy = py + ph * dy;
        float gw = pw * expf(dw), gh = ph * expf(dh);
        float c0 = gx - gw * 0.5f, c1 = gy - gh * 0.5f;
        float c2 = gx + gw * 0.5f, c3 = gy + gh * 0.5f;
        int b = r >> 9, n = r & 511;
        size_t m = (size_t)b * MCAND + (size_t)n * NCLS + tid;
        float o = (float)tid * 1536.0f;        // cls * 2*IMG
        float4 cb; cb.x = c0; cb.y = c1; cb.z = c2; cb.w = c3;
        ((float4*)cand_boxes)[m] = cb;
        float4 ob; ob.x = c0 + o; ob.y = c1 + o; ob.z = c2 + o; ob.w = c3 + o;
        ((float4*)boxes_off)[m] = ob;
        s0[m] = s0v;
    }
}

// ---------------------------------------------------------------------------
// NMS: one 1024-thread block per image; 160KB score array resident in CDNA5's
// 320KB LDS; 100 serial argmax+suppress rounds never touch HBM for scores.
// Output (f32): [counts(2)] [boxes(2*100*4)] [scores(2*100)] [cls(2*100)]
// ---------------------------------------------------------------------------
__global__ void nms_kernel(const float* __restrict__ cand_boxes,
                           const float* __restrict__ boxes_off,
                           const float* __restrict__ s0,
                           float* __restrict__ out)
{
    extern __shared__ float lds[];
    float* sc = lds;                          // MCAND floats
    float* rv = lds + MCAND;                  // 1024 floats
    int*   ri = (int*)(lds + MCAND + 1024);   // 1024 ints
    const int b = blockIdx.x, tid = threadIdx.x;

    for (int m = tid; m < MCAND; m += 1024) sc[m] = s0[(size_t)b * MCAND + m];
    __syncthreads();

    int count = 0;
    for (int it = 0; it < MAXDET; ++it) {
        float bv = -3.0e38f; int bi = 0;
        for (int m = tid; m < MCAND; m += 1024) {
            float v = sc[m];
            if (v > bv) { bv = v; bi = m; }
        }
        rv[tid] = bv; ri[tid] = bi;
        __syncthreads();
        for (int s = 512; s > 0; s >>= 1) {
            if (tid < s) {
                float ov = rv[tid + s]; int oi = ri[tid + s];
                if (ov > rv[tid] || (ov == rv[tid] && oi < ri[tid])) { rv[tid] = ov; ri[tid] = oi; }
            }
            __syncthreads();
        }
        int best = ri[0];
        float bscore = rv[0];
        bool ok = bscore > 0.0f;

        float4 bb = ((const float4*)boxes_off)[(size_t)b * MCAND + best];
        float ba = fmaxf(bb.z - bb.x, 0.f) * fmaxf(bb.w - bb.y, 0.f);
        if (ok) {
            for (int m = tid; m < MCAND; m += 1024) {
                float4 c = ((const float4*)boxes_off)[(size_t)b * MCAND + m];
                float ix1 = fmaxf(bb.x, c.x), iy1 = fmaxf(bb.y, c.y);
                float ix2 = fminf(bb.z, c.z), iy2 = fminf(bb.w, c.w);
                float inter = fmaxf(ix2 - ix1, 0.f) * fmaxf(iy2 - iy1, 0.f);
                float ca = fmaxf(c.z - c.x, 0.f) * fmaxf(c.w - c.y, 0.f);
                float iou = inter / (ba + ca - inter + 1e-6f);
                if (iou >= 0.5f) sc[m] = -1.0f;
            }
        }
        __syncthreads();
        if (tid == 0) {
            float ox = 0.f, oy = 0.f, ox2 = 0.f, oy2 = 0.f, scv = 0.f, cl = -1.0f;
            if (ok) {
                float4 cb = ((const float4*)cand_boxes)[(size_t)b * MCAND + best];
                ox = cb.x; oy = cb.y; ox2 = cb.z; oy2 = cb.w;
                scv = bscore; cl = (float)(best % NCLS);
                count++;
            }
            float* ob = out + 2 + ((size_t)b * MAXDET + it) * 4;
            ob[0] = ox; ob[1] = oy; ob[2] = ox2; ob[3] = oy2;
            out[2 + 2 * MAXDET * 4 + b * MAXDET + it] = scv;
            out[2 + 2 * MAXDET * 4 + 2 * MAXDET + b * MAXDET + it] = cl;
        }
    }
    if (tid == 0) out[b] = (float)count;
}

// ---------------------------------------------------------------------------
extern "C" void kernel_launch(void* const* d_in, const int* in_sizes, int n_in,
                              void* d_out, int out_size, void* d_ws, size_t ws_size,
                              hipStream_t stream) {
    const float* f0        = (const float*)d_in[0];
    const float* f1        = (const float*)d_in[1];
    const float* f2        = (const float*)d_in[2];
    const float* f3        = (const float*)d_in[3];
    const float* proposals = (const float*)d_in[4];
    const float* w1        = (const float*)d_in[5];
    const float* b1        = (const float*)d_in[6];
    const float* w2        = (const float*)d_in[7];
    const float* b2        = (const float*)d_in[8];
    const float* w_cls     = (const float*)d_in[9];
    const float* b_cls     = (const float*)d_in[10];
    const float* w_reg     = (const float*)d_in[11];
    const float* b_reg     = (const float*)d_in[12];
    float* out = (float*)d_out;

    char* base = (char*)d_ws;
    size_t off = 0;
    auto carve = [&](size_t bytes) -> void* {
        void* p = base + off;
        off = (off + bytes + 255) & ~(size_t)255;
        return p;
    };

    bf16*  xb    = (bf16*)carve((size_t)RROIS * DIN * 2);
    bf16*  w1b   = (bf16*)carve((size_t)DIN * FHID * 2);
    bf16*  w2b   = (bf16*)carve((size_t)FHID * FHID * 2);
    bf16*  wclsb = (bf16*)carve((size_t)FHID * NCLSP * 2);
    bf16*  wregb = (bf16*)carve((size_t)FHID * NREGP * 2);
    bf16*  h1    = (bf16*)carve((size_t)RROIS * FHID * 2);
    bf16*  h2    = (bf16*)carve((size_t)RROIS * FHID * 2);
    float* clsf  = (float*)carve((size_t)RROIS * NCLSP * 4);
    float* regf  = (float*)carve((size_t)RROIS * NREGP * 4);
    float* cboxes = (float*)carve((size_t)NB * MCAND * 4 * 4);
    float* oboxes = (float*)carve((size_t)NB * MCAND * 4 * 4);
    float* s0     = (float*)carve((size_t)NB * MCAND * 4);

    // 1) weight conversion fp32 -> bf16 (pad cls 81->128, reg 320->384 cols)
    {
        size_t t;
        t = (size_t)DIN * FHID;
        cvt_pad_bf16_kernel<<<(unsigned)((t + 255) / 256), 256, 0, stream>>>(w1, w1b, DIN, FHID, FHID);
        t = (size_t)FHID * FHID;
        cvt_pad_bf16_kernel<<<(unsigned)((t + 255) / 256), 256, 0, stream>>>(w2, w2b, FHID, FHID, FHID);
        t = (size_t)FHID * NCLSP;
        cvt_pad_bf16_kernel<<<(unsigned)((t + 255) / 256), 256, 0, stream>>>(w_cls, wclsb, FHID, NCLS + 1, NCLSP);
        t = (size_t)FHID * NREGP;
        cvt_pad_bf16_kernel<<<(unsigned)((t + 255) / 256), 256, 0, stream>>>(w_reg, wregb, FHID, NREG, NREGP);
    }

    // 2) RoI-Align -> bf16 activation matrix
    roi_align_kernel<<<RROIS, 256, 0, stream>>>(f0, f1, f2, f3, proposals, xb);

    // 3) GEMMs: block tile 64x128 (8 waves), async-LDS double buffered
    auto gemm = [&](const bf16* A, int lda, const bf16* W, int ldw,
                    const float* bias, int biasN,
                    float* oF, bf16* oB, int M, int N, int K, int relu) {
        int blocks = (M >> 6) * (N >> 7);
        wmma_gemm_bf16_kernel<<<blocks, 256, 0, stream>>>(A, lda, W, ldw, bias, biasN,
                                                          oF, oB, M, N, K, relu);
    };
    gemm(xb, DIN,  w1b,   FHID,  b1,    FHID,     nullptr, h1, RROIS, FHID,  DIN,  1);
    gemm(h1, FHID, w2b,   FHID,  b2,    FHID,     nullptr, h2, RROIS, FHID,  FHID, 1);
    gemm(h2, FHID, wclsb, NCLSP, b_cls, NCLS + 1, clsf, nullptr, RROIS, NCLSP, FHID, 0);
    gemm(h2, FHID, wregb, NREGP, b_reg, NREG,     regf, nullptr, RROIS, NREGP, FHID, 0);

    // 4) softmax + bbox decode + threshold
    smax_decode_kernel<<<RROIS, 128, 0, stream>>>(clsf, regf, proposals, cboxes, oboxes, s0);

    // 5) NMS, one block per image, scores resident in 320KB LDS
    size_t shmem = (size_t)(MCAND + 1024) * 4 + 1024 * 4;  // 172,032 B
    nms_kernel<<<NB, 1024, shmem, stream>>>(cboxes, oboxes, s0, out);
}
